// DiagonalGMMPosterior_64269890617659
// MI455X (gfx1250) — matrix operand order repaired and checked by reading, not compiled, not run on hardware
//
#include <hip/hip_runtime.h>
#include <hip/hip_bf16.h>
#include <math.h>

typedef __attribute__((ext_vector_type(16))) _Float16 v16h;
typedef __attribute__((ext_vector_type(8)))  float    v8f;

#define GB 16
#define GD 128
#define GN 16384
#define GK 64

// ws layout (dwords):
//   [0, 8192)     : W fragments, f16 pairs, index = ((((w12*4+t)*4+c)*32)+lane)*8 + j
//                   w12=0 -> -sinv (multiplies x^2), w12=1 -> 2*mu_n*sinv (multiplies x)
//   [8192, 8256)  : cvec[64]   = -mu_term + log_alpha - 0.5*sum(log_sigma)
//   [8256, 8320)  : norminv[64]

__global__ void gmm_prep_k(const float* __restrict__ mu,
                           const float* __restrict__ logsig,
                           const float* __restrict__ logalpha,
                           float* __restrict__ ws) {
  int k = threadIdx.x;
  if (k >= GK) return;
  const float* m  = mu + k * GD;
  const float* ls = logsig + k * GD;
  float n2 = 0.f, ssum = 0.f;
  for (int d = 0; d < GD; ++d) { n2 += m[d] * m[d]; ssum += ls[d]; }
  float ninv = 1.0f / fmaxf(sqrtf(n2), 1e-12f);
  float mt = 0.f;
  for (int d = 0; d < GD; ++d) {
    float mn = m[d] * ninv;
    mt += mn * mn * __expf(-ls[d]);
  }
  ws[8192 + k] = -mt + logalpha[k] - 0.5f * ssum;
  ws[8256 + k] = ninv;
}

// Bake W1/W2 into the per-lane VGPR layout of a 16x32 f16 A-fragment:
// lane L: M = L%16 (cluster within tile), hb = (L>=16)?8:0
// VGPR j<4  -> K(d) = c*32 + hb + 2j, 2j+1
// VGPR j>=4 -> K(d) = c*32 + hb + 16 + 2(j-4), +1
__global__ void gmm_prep_frag(const float* __restrict__ mu,
                              const float* __restrict__ logsig,
                              float* __restrict__ ws) {
  int idx = blockIdx.x * 256 + threadIdx.x;   // 0..8191
  int j   = idx & 7;
  int L   = (idx >> 3) & 31;
  int c   = (idx >> 8) & 3;
  int t   = (idx >> 10) & 3;
  int w12 = (idx >> 12) & 1;
  int k   = t * 16 + (L & 15);
  int hb  = (L >= 16) ? 8 : 0;
  int d0  = c * 32 + hb + ((j < 4) ? 2 * j : 16 + 2 * (j - 4));
  float ninv = ws[8256 + k];
  union { _Float16 h[2]; unsigned u; } p;
  for (int e = 0; e < 2; ++e) {
    int d = d0 + e;
    float sinv = __expf(-logsig[k * GD + d]);
    float v = (w12 == 0) ? (-sinv) : (2.0f * mu[k * GD + d] * ninv * sinv);
    p.h[e] = (_Float16)v;
  }
  ((unsigned*)ws)[idx] = p.u;
}

__global__ __launch_bounds__(128)
void gmm_main(const float* __restrict__ x,
              const float* __restrict__ ws_f,
              float* __restrict__ out) {
  __shared__ float lg[16 * 68];   // [n][k], pitch 68 dwords (bank stride 4)
  __shared__ float pmax[128];
  __shared__ float psum[128];
  __shared__ float maxn[16];
  __shared__ float invn[16];

  const int t    = threadIdx.x;
  const int w    = t >> 5;        // k-tile owned by this wave
  const int lane = t & 31;
  const int b    = blockIdx.y;
  const int nslab = blockIdx.x * 256;

  // Resident A-fragments for this wave's 16 clusters: 4 d-chunks x {W1,W2}
  const v16h* Wf = (const v16h*)ws_f;
  v16h aw1[4], aw2[4];
#pragma unroll
  for (int c = 0; c < 4; ++c) {
    aw1[c] = Wf[((0 * 4 + w) * 4 + c) * 32 + lane];
    aw2[c] = Wf[((1 * 4 + w) * 4 + c) * 32 + lane];
  }
  const int hb = (lane >> 4) * 8;      // C/D row offset for upper lane half
  float ck[8];
  {
    const float4* cp = (const float4*)(ws_f + 8192 + w * 16 + hb);
    float4 c0 = cp[0], c1 = cp[1];
    ck[0] = c0.x; ck[1] = c0.y; ck[2] = c0.z; ck[3] = c0.w;
    ck[4] = c1.x; ck[5] = c1.y; ck[6] = c1.z; ck[7] = c1.w;
  }

  const int tn = t & 15, tg = t >> 4;  // softmax work split: 8 k's per (n,group)
  const int ok = t >> 1, oh = t & 1;   // output: 2 threads per k-row

  for (int iter = 0; iter < 16; ++iter) {
    const int n0 = nslab + iter * 16;
    v8f acc = {0.f, 0.f, 0.f, 0.f, 0.f, 0.f, 0.f, 0.f};

#pragma unroll
    for (int c = 0; c < 4; ++c) {
      const int d = c * 32 + (lane >> 4) * 8;  // B-fragment: lane <-> n column, 16 d rows in table order
      const float* xp = x + ((size_t)b * GD + d) * GN + n0 + (lane & 15);
      // stream-prefetch next n-iteration's cacheline for this d row
      __builtin_prefetch((const void*)(xp + 16), 0, 1);
      float f[16];
#pragma unroll
      for (int i = 0; i < 16; ++i) f[i] = xp[(size_t)((i < 8) ? i : i + 8) * GN];
      v16h xb;
#pragma unroll
      for (int i = 0; i < 16; ++i) xb[i] = (_Float16)f[i];
      v16h xq = xb * xb;               // packed f16 squares (v_pk_mul_f16)
      acc = __builtin_amdgcn_wmma_f32_16x16x32_f16(false, aw2[c], false, xb,
                                                   (short)0, acc, false, false);
      acc = __builtin_amdgcn_wmma_f32_16x16x32_f16(false, aw1[c], false, xq,
                                                   (short)0, acc, false, false);
    }

    // deposit logits to LDS: n = lane&15, k = w*16 + hb + v
    {
      const int n = lane & 15;
      const int kb = w * 16 + hb;
#pragma unroll
      for (int v = 0; v < 8; ++v) lg[n * 68 + kb + v] = acc[v] + ck[v];
    }
    __syncthreads();

    // ---- softmax over K=64 for 16 columns, cooperatively ----
    {
      float m = -3.4e38f;
#pragma unroll
      for (int jj = 0; jj < 8; ++jj) m = fmaxf(m, lg[tn * 68 + tg * 8 + jj]);
      pmax[tg * 16 + tn] = m;
    }
    __syncthreads();
    if (t < 16) {
      float mm = pmax[t];
#pragma unroll
      for (int g = 1; g < 8; ++g) mm = fmaxf(mm, pmax[g * 16 + t]);
      maxn[t] = mm;
    }
    __syncthreads();
    {
      float mm = maxn[tn];
      float s = 0.f;
#pragma unroll
      for (int jj = 0; jj < 8; ++jj) {
        int li = tn * 68 + tg * 8 + jj;
        float e = __expf(lg[li] - mm);
        lg[li] = e;
        s += e;
      }
      psum[tg * 16 + tn] = s;
    }
    __syncthreads();
    if (t < 16) {
      float s = psum[t];
#pragma unroll
      for (int g = 1; g < 8; ++g) s += psum[g * 16 + t];
      invn[t] = 1.0f / s;
    }
    __syncthreads();

    // store posterior transposed: out[b][k][n0..n0+15], 32B contiguous per lane
    {
      const int nb = oh * 8;
      float r[8];
#pragma unroll
      for (int jj = 0; jj < 8; ++jj) r[jj] = lg[(nb + jj) * 68 + ok] * invn[nb + jj];
      float4* op = (float4*)(out + ((size_t)b * GK + ok) * GN + n0 + nb);
      op[0] = make_float4(r[0], r[1], r[2], r[3]);
      op[1] = make_float4(r[4], r[5], r[6], r[7]);
    }
    __syncthreads();
  }
}

extern "C" void kernel_launch(void* const* d_in, const int* in_sizes, int n_in,
                              void* d_out, int out_size, void* d_ws, size_t ws_size,
                              hipStream_t stream) {
  const float* x        = (const float*)d_in[0];
  const float* mu       = (const float*)d_in[1];
  const float* logsig   = (const float*)d_in[2];
  const float* logalpha = (const float*)d_in[3];
  float* out = (float*)d_out;
  float* ws  = (float*)d_ws;

  hipLaunchKernelGGL(gmm_prep_k,    dim3(1),            dim3(64),  0, stream,
                     mu, logsig, logalpha, ws);
  hipLaunchKernelGGL(gmm_prep_frag, dim3(32),           dim3(256), 0, stream,
                     mu, logsig, ws);
  hipLaunchKernelGGL(gmm_main,      dim3(GN / 256, GB), dim3(128), 0, stream,
                     x, ws, out);
}
